// AttentionAugmentation_19937238188595
// MI455X (gfx1250) — matrix operand order repaired and checked by reading, not compiled, not run on hardware
//
#include <hip/hip_runtime.h>
#include <hip/hip_bf16.h>

typedef __attribute__((ext_vector_type(16))) _Float16 v16h;
typedef __attribute__((ext_vector_type(8)))  float    v8f;

#define NH 8
#define WAVES 8
#define QPB 128   // query rows per workgroup (16 per wave)

// --- Fragment loaders for CDNA5 WMMA, contiguous k order (k = 16*(lane/16) + s) in both A and B ---

// A-matrix 16x32 f16, source is row-major [16][32] in LDS.
// lanes 0-15: row=lane, K = 0..15; lanes 16-31: row=lane-16, K = 16..31
__device__ inline v16h load_A_16x32(const _Float16* base, int lane) {
    int row = lane & 15;
    int kh  = (lane >> 4) << 4;          // 0 or 16
    const _Float16* p = base + row * 32 + kh;
    v16h f;
#pragma unroll
    for (int i = 0; i < 8; ++i) { f[i] = p[i]; f[8 + i] = p[8 + i]; }
    return f;
}

// B-matrix 32x16 f16 where B[k][n] = base[n*32 + k] (i.e. base is the 16xK tile row-major).
// lanes 0-15: col=lane, K=0..15; lanes 16-31: col=lane-16, K=16..31  (32 contiguous bytes)
__device__ inline v16h load_B_32x16(const _Float16* base, int lane) {
    int col = lane & 15;
    int kr  = (lane >> 4) << 4;          // 0 or 16
    const _Float16* p = base + col * 32 + kr;
    v16h f;
#pragma unroll
    for (int i = 0; i < 16; ++i) f[i] = p[i];
    return f;
}

__device__ inline v8f wmma_f16(v16h a, v16h b, v8f c) {
    return __builtin_amdgcn_wmma_f32_16x16x32_f16(false, a, false, b, (short)0, c, false, false);
}

__global__ __launch_bounds__(256)
void aug_attn_kernel(const float* __restrict__ inp,     // [8,32,32,768]
                     const float* __restrict__ relw,    // [63,32]
                     const float* __restrict__ relh,    // [63,32]
                     float* __restrict__ out)           // [8,32,32,256]
{
    __shared__ _Float16 Qbuf[QPB * 32];     // Q fragments source; reused as per-wave P buffers
    __shared__ _Float16 Kbuf[32 * 32];      // one 32-key chunk, row-major [key][d]
    __shared__ _Float16 Vt[32 * 32];        // V chunk transposed: [d][key]
    __shared__ _Float16 RW[QPB * 64];       // per-query relative-width bias (63 used, col 63 = 0)
    __shared__ _Float16 RH[QPB * 64];       // per-query relative-height bias
    __shared__ _Float16 RKW[64 * 32];       // padded key_rel_w (f16)
    __shared__ _Float16 RKH[64 * 32];       // padded key_rel_h (f16)

    const int tid  = threadIdx.x;
    const int lane = tid & 31;
    const int wv   = tid >> 5;
    const int qblk = blockIdx.x & 7;
    const int bh   = blockIdx.x >> 3;
    const int head = bh & (NH - 1);
    const int b    = bh >> 3;
    const int q0   = qblk * QPB;
    const float qscale = 0.17677669529663687f;   // 32^-0.5

    // ---- stage scaled Q block and padded relative keys ----
    for (int i = tid; i < QPB * 32; i += 256) {
        int row = i >> 5, d = i & 31;
        int q = q0 + row, h = q >> 5, w = q & 31;
        float v = inp[(((b * 32 + h) * 32 + w) * 768) + head * 32 + d] * qscale;
        Qbuf[i] = (_Float16)v;
    }
    for (int i = tid; i < 64 * 32; i += 256) {
        int r = i >> 5, d = i & 31;
        RKW[i] = (r < 63) ? (_Float16)relw[r * 32 + d] : (_Float16)0.f;
        RKH[i] = (r < 63) ? (_Float16)relh[r * 32 + d] : (_Float16)0.f;
    }
    __syncthreads();

    // persistent per-wave Q fragment (16 rows x 32)
    const v16h qfrag = load_A_16x32(Qbuf + wv * 16 * 32, lane);
    const int rowhalf = (lane >> 4) << 3;   // +0 or +8 on fragment row index

    // ---- relative-logit biases via WMMA: rw/rh[m][j] = q[m] . rel_k[j] ----
#pragma unroll
    for (int n0 = 0; n0 < 64; n0 += 16) {
        v8f cw = {}; cw = wmma_f16(qfrag, load_B_32x16(RKW + n0 * 32, lane), cw);
        v8f ch = {}; ch = wmma_f16(qfrag, load_B_32x16(RKH + n0 * 32, lane), ch);
#pragma unroll
        for (int r = 0; r < 8; ++r) {
            int m = wv * 16 + r + rowhalf;
            RW[m * 64 + n0 + (lane & 15)] = (_Float16)cw[r];
            RH[m * 64 + n0 + (lane & 15)] = (_Float16)ch[r];
        }
    }
    // RW/RH rows are produced and consumed by the same wave; no barrier needed here.

    // ---- flash-attention state ----
    float rm[8], rl[8];
#pragma unroll
    for (int r = 0; r < 8; ++r) { rm[r] = -1e30f; rl[r] = 0.f; }
    v8f O0 = {}, O1 = {};

    _Float16* Pb = Qbuf + wv * 16 * 32;     // per-wave P scratch (reuses Q buffer)

    for (int kc = 0; kc < 32; ++kc) {       // chunk kc == image key-row h2
        __syncthreads();                    // previous chunk fully consumed
        for (int i = tid; i < 32 * 32; i += 256) {
            int key = i >> 5, d = i & 31;
            const float* src = inp + (((b * 32 + kc) * 32 + key) * 768) + head * 32 + d;
            Kbuf[i]          = (_Float16)src[256];   // K
            Vt[d * 32 + key] = (_Float16)src[512];   // V transposed
        }
        __syncthreads();

        // S = Q x K^T for 16 rows x 32 keys
        v8f s0 = {}; s0 = wmma_f16(qfrag, load_B_32x16(Kbuf, lane), s0);
        v8f s1 = {}; s1 = wmma_f16(qfrag, load_B_32x16(Kbuf + 16 * 32, lane), s1);

        const int c0 = lane & 15;           // fragment column
#pragma unroll
        for (int r = 0; r < 8; ++r) {
            int mt = r + rowhalf;           // row within 16-row tile
            int m  = wv * 16 + mt;
            int q  = q0 + m, h = q >> 5, w = q & 31;
            float bhh = (float)RH[m * 64 + (kc - h + 31)];
            float v0 = s0[r] + bhh + (float)RW[m * 64 + (c0      - w + 31)];
            float v1 = s1[r] + bhh + (float)RW[m * 64 + (c0 + 16 - w + 31)];

            // row max over the 32 keys (16-lane halves == fragment rows)
            float t = fmaxf(v0, v1);
            t = fmaxf(t, __shfl_xor(t, 1));
            t = fmaxf(t, __shfl_xor(t, 2));
            t = fmaxf(t, __shfl_xor(t, 4));
            t = fmaxf(t, __shfl_xor(t, 8));

            float nm = fmaxf(rm[r], t);
            float sc = __expf(rm[r] - nm);
            float e0 = __expf(v0 - nm);
            float e1 = __expf(v1 - nm);
            float ss = e0 + e1;
            ss += __shfl_xor(ss, 1);
            ss += __shfl_xor(ss, 2);
            ss += __shfl_xor(ss, 4);
            ss += __shfl_xor(ss, 8);
            rl[r] = rl[r] * sc + ss;
            rm[r] = nm;
            O0[r] *= sc;
            O1[r] *= sc;
            Pb[mt * 32 + c0]      = (_Float16)e0;   // C-layout -> A-layout via LDS
            Pb[mt * 32 + 16 + c0] = (_Float16)e1;
        }

        // O += P x V   (same-wave LDS RAW: DS ops are in-order within a wave)
        v16h pa = load_A_16x32(Pb, lane);
        O0 = wmma_f16(pa, load_B_32x16(Vt, lane), O0);
        O1 = wmma_f16(pa, load_B_32x16(Vt + 16 * 32, lane), O1);
    }

    // ---- normalize and write out[b,h,w, head*32 + d] ----
    const int c0 = lane & 15;
#pragma unroll
    for (int r = 0; r < 8; ++r) {
        int m = wv * 16 + r + rowhalf;
        int q = q0 + m, h = q >> 5, w = q & 31;
        float inv = 1.0f / rl[r];
        float* o = out + (((b * 32 + h) * 32 + w) * 256) + head * 32;
        o[c0]      = O0[r] * inv;
        o[c0 + 16] = O1[r] * inv;
    }
}

extern "C" void kernel_launch(void* const* d_in, const int* in_sizes, int n_in,
                              void* d_out, int out_size, void* d_ws, size_t ws_size,
                              hipStream_t stream) {
    const float* inputs = (const float*)d_in[0];
    const float* relw   = (const float*)d_in[1];
    const float* relh   = (const float*)d_in[2];
    float* out          = (float*)d_out;
    (void)in_sizes; (void)n_in; (void)out_size; (void)d_ws; (void)ws_size;

    dim3 grid(8 * NH * (1024 / QPB));   // 512 workgroups: (b, head, query-block)
    dim3 block(256);                    // 8 waves of 32
    aug_attn_kernel<<<grid, block, 0, stream>>>(inputs, relw, relh, out);
}
